// Adex_EI_80693845557793
// MI455X (gfx1250) — matrix operand order repaired and checked by reading, not compiled, not run on hardware
//
#include <hip/hip_runtime.h>
#include <stdint.h>

// ---------------- problem dims ----------------
#define NB   512                 // batch
#define NIN  2048                // input features
#define NU   4096                // units
#define KTOT (NIN + NU)          // 6144 fused K
#define BM   64                  // block tile M
#define BN   128                 // block tile N
#define BK   32                  // K step (bf16 wmma K)

// ---------------- AdEx constants ----------------
#define F_DT_GL_C    (30.0f / 281.0f)      // DT*GL/CAP
#define F_DT_CAP     (1.0f / 281.0f)       // DT/CAP
#define F_DT_TAUW    (1.0f / 144.0f)       // DT/TAUW
#define F_DT_A_TAUW  (4.0f / 144.0f)       // DT*A/TAUW
#define F_BW         0.0805f
#define F_THR        (-50.4f)
#define F_EL         (-70.6f)
#define F_VRESET     (-70.6f)
#define F_EXPCLIP    281.0f                // 30/DT_GL_C
#define F_INV_THR_EL (1.0f / 20.2f)        // 1/(THR-EL)

typedef __attribute__((ext_vector_type(8)))  __bf16 v8bf;
typedef __attribute__((ext_vector_type(16))) __bf16 v16bf;
typedef __attribute__((ext_vector_type(8)))  float  v8f;
typedef __attribute__((ext_vector_type(4)))  unsigned int tdm_v4u;
typedef __attribute__((ext_vector_type(8)))  int          tdm_v8i;
typedef __attribute__((ext_vector_type(4)))  int          tdm_v4i;

// fp32 -> bf16 round-to-nearest-even (bit pattern)
__device__ __forceinline__ unsigned short f2bf(float f) {
  unsigned int u = __float_as_uint(f);
  u += 0x7FFFu + ((u >> 16) & 1u);
  return (unsigned short)(u >> 16);
}

// ---------------------------------------------------------------------------
// Prep 1: Acat[m][k] = bf16( k<NIN ? inputs[m][k] : z[m][k-NIN] )
// ---------------------------------------------------------------------------
__global__ void adex_pack_A(const float* __restrict__ inputs,
                            const float* __restrict__ z,
                            unsigned short* __restrict__ Acat) {
  int idx = blockIdx.x * blockDim.x + threadIdx.x;
  if (idx >= NB * KTOT) return;
  int m = idx / KTOT;
  int k = idx - m * KTOT;
  float val = (k < NIN) ? inputs[m * NIN + k] : z[(size_t)m * NU + (k - NIN)];
  Acat[idx] = f2bf(val);
}

// ---------------------------------------------------------------------------
// Prep 2: WT[n][k] = bf16( k<NIN ? Win[k][n] : Wrec[k-NIN][n] ), diag zeroed.
// LDS-tiled 32x32 transpose: coalesced on both sides.
// ---------------------------------------------------------------------------
__global__ void adex_pack_WT(const float* __restrict__ Win,
                             const float* __restrict__ Wrec,
                             unsigned short* __restrict__ WT) {
  __shared__ unsigned short tile[32][33];
  const int k0 = blockIdx.x * 32;
  const int n0 = blockIdx.y * 32;
  const int tx = threadIdx.x;      // 0..31
  const int ty = threadIdx.y;      // 0..7
#pragma unroll
  for (int i = 0; i < 32; i += 8) {
    int k = k0 + ty + i;
    int n = n0 + tx;
    float val;
    if (k < NIN) {
      val = Win[(size_t)k * NU + n];
    } else {
      int kr = k - NIN;
      val = (kr == n) ? 0.0f : Wrec[(size_t)kr * NU + n];   // zero autapses
    }
    tile[ty + i][tx] = f2bf(val);
  }
  __syncthreads();
#pragma unroll
  for (int i = 0; i < 32; i += 8) {
    WT[(size_t)(n0 + ty + i) * KTOT + (k0 + tx)] = tile[tx][ty + i];
  }
}

// ---------------------------------------------------------------------------
// TDM: issue one 2D tile load (rows x BK bf16, row stride KTOT) into LDS.
// D# packed per CDNA5 ISA sec. 8.3/8.4 (group0 128b, group1 256b, grp2/3 = 0).
// amdgpu-toolchain (clang-23) 6-arg builtin form.
// ---------------------------------------------------------------------------
__device__ __forceinline__ void tdm_load_tile(unsigned lds_addr,
                                              const unsigned short* gptr,
                                              unsigned tile_rows,
                                              unsigned tensor_rows) {
  unsigned long long ga = (unsigned long long)(uintptr_t)gptr;
  tdm_v4u g0 = {
      1u,                                   // count=1, user descriptor
      lds_addr,                             // lds_addr [63:32]
      (unsigned)ga,                         // global_addr low
      (unsigned)((ga >> 32) & 0x1FFFFFFull) | (2u << 30)  // addr[56:32] | type=2
  };
  tdm_v8i g1 = {
      (int)(1u << 16),                                          // data_size=2B
      (int)(((unsigned)KTOT & 0xFFFFu) << 16),                  // tensor_dim0 lo16
      (int)(((unsigned)KTOT >> 16) | ((tensor_rows & 0xFFFFu) << 16)),
      (int)((tensor_rows >> 16) | ((unsigned)BK << 16)),        // tile_dim0=BK
      (int)tile_rows,                                           // tile_dim1
      (int)KTOT,                                                // dim0 stride lo32
      0, 0
  };
  tdm_v4i zg4 = {0, 0, 0, 0};
  tdm_v8i zg8 = {0, 0, 0, 0, 0, 0, 0, 0};
  __builtin_amdgcn_tensor_load_to_lds(g0, g1, zg4, zg4, zg8, 0);
}

// ---------------------------------------------------------------------------
// Fused GEMM (bf16 WMMA, fp32 acc) + AdEx pointwise epilogue.
// Block: 256 threads = 8 waves arranged 2(M) x 4(N); wave tile 32x32.
// ---------------------------------------------------------------------------
__global__ __launch_bounds__(256)
void adex_gemm_fused(const unsigned short* __restrict__ Acat,
                     const unsigned short* __restrict__ WT,
                     const float* __restrict__ vin,
                     const int*   __restrict__ rin,
                     const float* __restrict__ win,
                     const float* __restrict__ zin,
                     float* __restrict__ outV,
                     float* __restrict__ outZ,
                     float* __restrict__ outW,
                     int*   __restrict__ outR) {
  __shared__ __align__(16) unsigned short sA[2][BM * BK];   // 2 x 4KB
  __shared__ __align__(16) unsigned short sB[2][BN * BK];   // 2 x 8KB

  const int tid   = threadIdx.x;
  const int lane  = tid & 31;
  const int wave  = tid >> 5;
  const int waveM = wave >> 2;     // 0..1
  const int waveN = wave & 3;      // 0..3
  const int bM = blockIdx.y * BM;
  const int bN = blockIdx.x * BN;

  const unsigned ldsA = (unsigned)(size_t)(void*)&sA[0][0];
  const unsigned ldsB = (unsigned)(size_t)(void*)&sB[0][0];

  v8f acc[2][2] = {};

  // prologue: load K-tile 0 into buffer 0
  if (wave == 0) {
    tdm_load_tile(ldsA, Acat + (size_t)bM * KTOT, BM, NB);
    tdm_load_tile(ldsB, WT + (size_t)bN * KTOT, BN, NU);
  }

  const int KT = KTOT / BK;        // 192
  const int koffA = (lane & 16) ? 16 : 0;   // A: lanes 16-31 hold K 16..31
  const int koffB = (lane & 16) ? 16 : 0;   // B: lanes 16-31 hold K 16..31
  const int l15 = lane & 15;

  for (int kk = 0; kk < KT; ++kk) {
    const int cur = kk & 1;
    if (wave == 0) __builtin_amdgcn_s_wait_tensorcnt(0);  // buf[cur] landed
    __syncthreads();

    // prefetch next K-tile into the other buffer (its readers passed the
    // previous iteration's trailing barrier)
    if (wave == 0 && (kk + 1) < KT) {
      const size_t k0 = (size_t)(kk + 1) * BK;
      const int nxt = cur ^ 1;
      tdm_load_tile(ldsA + (unsigned)(nxt * BM * BK * 2),
                    Acat + (size_t)bM * KTOT + k0, BM, NB);
      tdm_load_tile(ldsB + (unsigned)(nxt * BN * BK * 2),
                    WT + (size_t)bN * KTOT + k0, BN, NU);
    }

    // build fragments in contiguous k order (k = 16*(lane/16) + s), same for A and B
    v16bf afrag[2], bfrag[2];
#pragma unroll
    for (int mt = 0; mt < 2; ++mt) {
      const unsigned short* p = &sA[cur][(waveM * 32 + mt * 16 + l15) * BK];
      v8bf lo = *(const v8bf*)(p + koffA);
      v8bf hi = *(const v8bf*)(p + koffA + 8);
      afrag[mt] = __builtin_shufflevector(lo, hi, 0, 1, 2, 3, 4, 5, 6, 7,
                                          8, 9, 10, 11, 12, 13, 14, 15);
    }
#pragma unroll
    for (int nt = 0; nt < 2; ++nt) {
      bfrag[nt] = *(const v16bf*)&sB[cur][(waveN * 32 + nt * 16 + l15) * BK + koffB];
    }
#pragma unroll
    for (int mt = 0; mt < 2; ++mt)
#pragma unroll
      for (int nt = 0; nt < 2; ++nt)
        acc[mt][nt] = __builtin_amdgcn_wmma_f32_16x16x32_bf16(
            false, afrag[mt], false, bfrag[nt],
            (short)0, acc[mt][nt], false, false);

    __syncthreads();
  }

  // -------- fused AdEx epilogue on C fragments --------
  // C layout: VGPR j -> row (j or j+8), lane&15 -> col, lane>=16 -> +8 rows.
  const int rowHi = (lane & 16) ? 8 : 0;
#pragma unroll
  for (int mt = 0; mt < 2; ++mt) {
#pragma unroll
    for (int nt = 0; nt < 2; ++nt) {
      const int mBase = bM + waveM * 32 + mt * 16 + rowHi;
      const int n     = bN + waveN * 32 + nt * 16 + l15;
#pragma unroll
      for (int j = 0; j < 8; ++j) {
        const size_t idx = (size_t)(mBase + j) * NU + n;
        const float it = acc[mt][nt][j];
        const float vv = vin[idx];
        const int   rr = rin[idx];
        const float ww = win[idx];
        const float zz = zin[idx];

        float ex = __expf((vv - F_THR) * 0.5f);          // (v-THR)/DELTAT
        ex = fminf(ex, F_EXPCLIP);
        float nv = vv - F_DT_GL_C * (vv - F_EL)
                 + F_DT_GL_C * 2.0f * ex
                 + (it - ww) * F_DT_CAP;
        nv = (zz > 0.5f) ? F_VRESET : nv;
        float nw = ww - F_DT_TAUW * ww + F_DT_A_TAUW * (vv - F_EL) + F_BW * zz;
        float vs = (nv - F_THR) * F_INV_THR_EL;
        float nz = (vs > 0.0f) ? 1.0f : 0.0f;
        if (rr > 0) nz = 0.0f;
        int nr = rr - 1 + (int)(nz * 2.0f);
        nr = nr < 0 ? 0 : (nr > 2 ? 2 : nr);

        outV[idx] = nv;
        outZ[idx] = nz;
        outW[idx] = nw;
        outR[idx] = nr;
      }
    }
  }
}

// ---------------------------------------------------------------------------
extern "C" void kernel_launch(void* const* d_in, const int* in_sizes, int n_in,
                              void* d_out, int out_size, void* d_ws, size_t ws_size,
                              hipStream_t stream) {
  (void)in_sizes; (void)n_in; (void)out_size; (void)ws_size;

  const float* inputs = (const float*)d_in[0];
  const float* v      = (const float*)d_in[1];
  const int*   r      = (const int*)  d_in[2];
  const float* w      = (const float*)d_in[3];
  const float* z      = (const float*)d_in[4];
  const float* Win    = (const float*)d_in[5];
  const float* Wrec   = (const float*)d_in[6];

  // workspace: Acat bf16 [NB][KTOT] then WT bf16 [NU][KTOT]  (~56.6 MB)
  unsigned short* Acat = (unsigned short*)d_ws;
  unsigned short* WT   = Acat + (size_t)NB * KTOT;

  const size_t BU = (size_t)NB * NU;
  float* outV = (float*)d_out;
  float* outZ = outV + BU;
  float* outW = outZ + BU;
  int*   outR = (int*)(outW + BU);

  adex_pack_A<<<(NB * KTOT + 255) / 256, 256, 0, stream>>>(inputs, z, Acat);
  adex_pack_WT<<<dim3(KTOT / 32, NU / 32), dim3(32, 8), 0, stream>>>(Win, Wrec, WT);
  adex_gemm_fused<<<dim3(NU / BN, NB / BM), 256, 0, stream>>>(
      Acat, WT, v, r, w, z, outV, outZ, outW, outR);
}
